// BidirectionalCrossAttention_30528627540101
// MI455X (gfx1250) — matrix operand order repaired and checked by reading, not compiled, not run on hardware
//
#include <hip/hip_runtime.h>
#include <hip/hip_bf16.h>
#include <math.h>

#define DEV __device__ __forceinline__

typedef __bf16 bf16_t;
typedef __attribute__((ext_vector_type(16))) __bf16 v16bf;
typedef __attribute__((ext_vector_type(8)))  __bf16 bf16x8;
typedef __attribute__((ext_vector_type(8)))  float  v8f;
typedef __attribute__((ext_vector_type(4)))  unsigned int u32x4;
typedef __attribute__((ext_vector_type(8)))  int i32x8;
typedef __attribute__((ext_vector_type(4)))  int i32x4;

static constexpr int B_    = 2;
static constexpr int SEQ   = 2048;
static constexpr int DM    = 256;
static constexpr int H     = 8;
static constexpr int DH    = 64;
static constexpr int INNER = 512;   // H*DH
static constexpr int MROWS = B_ * SEQ;      // 4096
static constexpr int CROWS = B_ * 2 * SEQ;  // 8192 combined rows

DEV int lane_id() { return (int)(threadIdx.x & 31); }

// A fragment (16x32 bf16, row-major source): lane = {row = l&15, half = l>>4};
// element e<8 -> K = 8*half + e ; element e>=8 -> K = 16 + 8*half + (e-8).
DEV v16bf load_a16x32(const bf16_t* A, int lda, int row0, int k0) {
  const int l = lane_id();
  const int r = l & 15, hf = l >> 4;
  const bf16_t* p = A + (size_t)(row0 + r) * lda + k0 + hf * 8;
  bf16x8 lo = *reinterpret_cast<const bf16x8*>(p);
  bf16x8 hi = *reinterpret_cast<const bf16x8*>(p + 16);
  v16bf f;
#pragma unroll
  for (int t = 0; t < 8; ++t) { f[t] = lo[t]; f[t + 8] = hi[t]; }
  return f;
}

// B fragment (32x16 bf16) from TRANSPOSED row-major BT[N, K]:
// lane = {col = l&15, half = l>>4}; element e<8 -> K = 8*half + e ; element e>=8 -> K = 16 + 8*half + (e-8).
DEV v16bf load_bT16x32(const bf16_t* BT, int ldb, int n0, int k0) {
  const int l = lane_id();
  const int c = l & 15, hf = l >> 4;
  const bf16_t* p = BT + (size_t)(n0 + c) * ldb + k0 + hf * 8;
  bf16x8 lo = *reinterpret_cast<const bf16x8*>(p);
  bf16x8 hi = *reinterpret_cast<const bf16x8*>(p + 16);
  v16bf f;
#pragma unroll
  for (int t = 0; t < 8; ++t) { f[t] = lo[t]; f[t + 8] = hi[t]; }
  return f;
}

DEV v8f wmma_bf16(v16bf a, v16bf b, v8f c) {
  return __builtin_amdgcn_wmma_f32_16x16x32_bf16(false, a, false, b, (short)0, c,
                                                 false, false);
}

// LDS byte offset of a __shared__ object (AS3 pointers are offsets into LDS)
DEV unsigned lds_off(const void* p) {
  return (unsigned)(uintptr_t)(__attribute__((address_space(3))) const void*)p;
}

// Issue a 2-D TDM tile load: global (row-major, row pitch stride0 elements,
// tile0 x tile1 bf16 elements) -> LDS at lds_byte_off, with per-row padding so
// the LDS row stride becomes tile0*2 + 16 bytes (bank stagger).
// pad_icode: interval = 2^(pad_icode+1) dwords (== one row);  pad 4 dwords (16B).
DEV void tdm_load_2d(unsigned lds_byte_off, const void* gaddr,
                     unsigned tile0, unsigned tile1, unsigned stride0,
                     unsigned pad_icode) {
  uint64_t ga = (uint64_t)(uintptr_t)gaddr;
  u32x4 g0;
  g0[0] = 1u;                                   // count=1 user descriptor
  g0[1] = lds_byte_off;                         // D#.lds_addr
  g0[2] = (unsigned)(ga & 0xFFFFFFFFu);         // global_addr[31:0]
  g0[3] = (unsigned)((ga >> 32) & 0x01FFFFFFu)  // global_addr[56:32]
          | (2u << 30);                         // type = 2 ("image")
  const unsigned td0 = tile0, td1 = tile1;      // tensor dims == tile dims
  i32x8 g1;
  g1[0] = (int)((1u << 16)                      // data_size = 1 (2 bytes)
                | (1u << 20)                    // pad_enable
                | (pad_icode << 22)             // pad_interval (one row)
                | (3u << 25));                  // pad_amount = 4 dwords (16 B)
  g1[1] = (int)((td0 & 0xFFFFu) << 16);                       // tensor_dim0 lo
  g1[2] = (int)((td0 >> 16) | ((td1 & 0xFFFFu) << 16));       // td0 hi | td1 lo
  g1[3] = (int)((td1 >> 16) | (tile0 << 16));                 // td1 hi | tile_dim0
  g1[4] = (int)(tile1 & 0xFFFFu);                             // tile_dim1 (tile2=0)
  g1[5] = (int)stride0;                                       // tensor_dim0_stride lo
  g1[6] = 0;                                                  // stride0 hi | stride1 lo
  g1[7] = 0;
  i32x4 z4 = {0, 0, 0, 0};
  i32x8 z8 = {0, 0, 0, 0, 0, 0, 0, 0};
  // 6-arg form (clang-23 / therock-10.0 headers): g0, g1, g2, g3, g4, cpol
  __builtin_amdgcn_tensor_load_to_lds(g0, g1, z4, z4, z8, 0);
}

// ---------------------------------------------------------------------------
__global__ __launch_bounds__(256) void f2bf_kernel(const float* __restrict__ src,
                                                   bf16_t* __restrict__ dst, int n) {
  int i = blockIdx.x * blockDim.x + threadIdx.x;
  if (i < n) dst[i] = (bf16_t)src[i];
}

__global__ __launch_bounds__(256) void wtrans_kernel(const float* __restrict__ W,
                                                     bf16_t* __restrict__ WT,
                                                     int K, int N) {
  int i = blockIdx.x * blockDim.x + threadIdx.x;
  if (i < K * N) {
    int k = i / N, n = i - k * N;
    WT[(size_t)n * K + k] = (bf16_t)W[i];
  }
}

// ---------------------------------------------------------------------------
// GEMM: C[M,N] = A[M,K](bf16 rm) x BT[N,K](bf16 rm)^T (+bias).  MODE as before.
template <int MODE>
__global__ __launch_bounds__(256)
void gemm_bf16_kernel(const bf16_t* __restrict__ A, const bf16_t* __restrict__ BT,
                      const float* __restrict__ bias, void* __restrict__ outp,
                      int N, int K, int rows_in_log2, int rows_out, int row_off) {
  const int bm = blockIdx.y * 128;
  const int bn = blockIdx.x * 64;
  const int w  = threadIdx.x >> 5;
  const int wm = w & 3, wn = w >> 2;
  const int row0 = bm + wm * 32;
  const int col0 = bn + wn * 32;

  v8f acc00 = {}, acc01 = {}, acc10 = {}, acc11 = {};
  for (int k0 = 0; k0 < K; k0 += 32) {
    v16bf a0 = load_a16x32(A, K, row0, k0);
    v16bf a1 = load_a16x32(A, K, row0 + 16, k0);
    v16bf b0 = load_bT16x32(BT, K, col0, k0);
    v16bf b1 = load_bT16x32(BT, K, col0 + 16, k0);
    acc00 = wmma_bf16(a0, b0, acc00);
    acc01 = wmma_bf16(a0, b1, acc01);
    acc10 = wmma_bf16(a1, b0, acc10);
    acc11 = wmma_bf16(a1, b1, acc11);
  }

  const int l = lane_id(), hf = l >> 4, nl = l & 15;
#pragma unroll
  for (int tm = 0; tm < 2; ++tm) {
#pragma unroll
    for (int tn = 0; tn < 2; ++tn) {
      v8f a = (tm == 0) ? ((tn == 0) ? acc00 : acc01) : ((tn == 0) ? acc10 : acc11);
#pragma unroll
      for (int e = 0; e < 8; ++e) {
        int gr = row0 + tm * 16 + e + 8 * hf;
        int gc = col0 + tn * 16 + nl;
        float v = a[e];
        if (bias) v += bias[gc];
        if (MODE == 0) {
          int bb = gr >> rows_in_log2;
          int rr = gr & ((1 << rows_in_log2) - 1);
          ((bf16_t*)outp)[((size_t)bb * rows_out + row_off + rr) * N + gc] = (bf16_t)v;
        } else if (MODE == 1) {
          int bb = gr >> 11, rr = gr & (SEQ - 1);
          int hh = gc >> 6, dd = gc & (DH - 1);
          ((bf16_t*)outp)[(((size_t)(bb * H + hh)) * DH + dd) * SEQ + rr] = (bf16_t)v;
        } else {
          ((float*)outp)[(size_t)gr * N + gc] = v;
        }
      }
    }
  }
}

// ---------------------------------------------------------------------------
// Flash attention with TDM double-buffered K/V tiles in LDS.
//   Q, Kmat: [B*SEQ, INNER] bf16 rm; VT: [B, H, DH, SEQ] bf16; O like Q.
// K tile in LDS: 128 rows x 64 el, padded stride 72 el (144 B).
// V tile in LDS:  64 rows x 128 el, padded stride 136 el (272 B).
static constexpr int LDSP  = 136;  // P / V-tile padded stride (elements)
static constexpr int KLDSP = 72;   // K-tile padded stride (elements)

__global__ __launch_bounds__(256)
void flash_attn_kernel(const bf16_t* __restrict__ Q, const bf16_t* __restrict__ Kmat,
                       const bf16_t* __restrict__ VT, bf16_t* __restrict__ O,
                       float scale) {
  __shared__ __align__(16) bf16_t Kt[2][128 * KLDSP];
  __shared__ __align__(16) bf16_t Vt[2][64 * LDSP];
  __shared__ __align__(16) bf16_t Pl[8 * 16 * LDSP];

  const int bh = blockIdx.y;
  const int b  = bh >> 3, h = bh & 7;
  const int i0 = blockIdx.x * 128;
  const int w  = threadIdx.x >> 5;
  const int l  = lane_id(), hf = l >> 4, nl = l & 15;
  const int qrow0 = b * SEQ + i0 + w * 16;
  const int kcol  = h * DH;
  const bf16_t* Kb = Kmat + (size_t)b * SEQ * INNER + kcol;  // [SEQ, 64] view, pitch INNER
  const bf16_t* Vb = VT + ((size_t)(b * H + h)) * DH * SEQ;  // [DH, SEQ] view
  bf16_t* Pw = &Pl[w * 16 * LDSP];

  // Loop-invariant Q fragments (K-dim = DH = 64 -> two 16x32 frags)
  v16bf aQ0 = load_a16x32(Q, INNER, qrow0, kcol);
  v16bf aQ1 = load_a16x32(Q, INNER, qrow0, kcol + 32);

  // Prologue: DMA first K/V tiles into buffer 0 (wave 0 issues; TENSORcnt gates)
  if (w == 0) {
    tdm_load_2d(lds_off(&Kt[0][0]), Kb, DH, 128, INNER, 4);      // 128B rows
    tdm_load_2d(lds_off(&Vt[0][0]), Vb, 128, DH, SEQ, 5);        // 256B rows
    __builtin_amdgcn_s_wait_tensorcnt(0);
  }
  __syncthreads();

  v8f o0 = {}, o1 = {}, o2 = {}, o3 = {};
  float mi[8], li[8];
#pragma unroll
  for (int e = 0; e < 8; ++e) { mi[e] = -INFINITY; li[e] = 0.f; }

  int cur = 0;
  for (int j0 = 0; j0 < SEQ; j0 += 128) {
    // Prefetch next tiles into the other buffer (overlaps with WMMA below)
    if (w == 0 && j0 + 128 < SEQ) {
      tdm_load_2d(lds_off(&Kt[cur ^ 1][0]), Kb + (size_t)(j0 + 128) * INNER,
                  DH, 128, INNER, 4);
      tdm_load_2d(lds_off(&Vt[cur ^ 1][0]), Vb + (j0 + 128), 128, DH, SEQ, 5);
    }
    const bf16_t* KtC = &Kt[cur][0];
    const bf16_t* VtC = &Vt[cur][0];

    // S = Q . K^T (fragments from LDS; shared by all 8 waves)
    v8f s[8];
#pragma unroll
    for (int jt = 0; jt < 8; ++jt) {
      v8f c = {};
      c = wmma_bf16(aQ0, load_bT16x32(KtC, KLDSP, jt * 16, 0), c);
      c = wmma_bf16(aQ1, load_bT16x32(KtC, KLDSP, jt * 16, 32), c);
      s[jt] = c;
    }

    // online softmax: row max over 8 tiles then 16 lanes within half
    float corr[8];
#pragma unroll
    for (int e = 0; e < 8; ++e) {
      float mx = s[0][e];
#pragma unroll
      for (int jt = 1; jt < 8; ++jt) mx = fmaxf(mx, s[jt][e]);
      mx = fmaxf(mx, __shfl_xor(mx, 1));
      mx = fmaxf(mx, __shfl_xor(mx, 2));
      mx = fmaxf(mx, __shfl_xor(mx, 4));
      mx = fmaxf(mx, __shfl_xor(mx, 8));
      float mn = fmaxf(mi[e], mx);
      corr[e]  = __expf(scale * (mi[e] - mn));
      mi[e]    = mn;
    }

    // P = exp(scale*(S - m)) -> wave-private LDS tile (A-frag layout source)
    float rs[8];
#pragma unroll
    for (int e = 0; e < 8; ++e) rs[e] = 0.f;
#pragma unroll
    for (int jt = 0; jt < 8; ++jt) {
#pragma unroll
      for (int e = 0; e < 8; ++e) {
        float pv = __expf(scale * (s[jt][e] - mi[e]));
        rs[e] += pv;
        Pw[(e + 8 * hf) * LDSP + jt * 16 + nl] = (bf16_t)pv;
      }
    }
#pragma unroll
    for (int e = 0; e < 8; ++e) {
      float r = rs[e];
      r += __shfl_xor(r, 1);
      r += __shfl_xor(r, 2);
      r += __shfl_xor(r, 4);
      r += __shfl_xor(r, 8);
      li[e] = li[e] * corr[e] + r;
      float cc = corr[e];
      o0[e] *= cc; o1[e] *= cc; o2[e] *= cc; o3[e] *= cc;
    }

    // O += P @ V (P A-frags from LDS; V B-frags from LDS tile)
#pragma unroll
    for (int jc = 0; jc < 4; ++jc) {
      v16bf aP = load_a16x32(Pw, LDSP, 0, jc * 32);
      o0 = wmma_bf16(aP, load_bT16x32(VtC, LDSP, 0,  jc * 32), o0);
      o1 = wmma_bf16(aP, load_bT16x32(VtC, LDSP, 16, jc * 32), o1);
      o2 = wmma_bf16(aP, load_bT16x32(VtC, LDSP, 32, jc * 32), o2);
      o3 = wmma_bf16(aP, load_bT16x32(VtC, LDSP, 48, jc * 32), o3);
    }

    // Next-tile DMA must be complete and all waves done with this buffer
    if (w == 0) __builtin_amdgcn_s_wait_tensorcnt(0);
    __syncthreads();
    cur ^= 1;
  }

  // normalize and write bf16 head slice
#pragma unroll
  for (int e = 0; e < 8; ++e) {
    float inv = 1.f / li[e];
    size_t rb = (size_t)(qrow0 + e + 8 * hf) * INNER + kcol;
    O[rb + 0  + nl] = (bf16_t)(o0[e] * inv);
    O[rb + 16 + nl] = (bf16_t)(o1[e] * inv);
    O[rb + 32 + nl] = (bf16_t)(o2[e] * inv);
    O[rb + 48 + nl] = (bf16_t)(o3[e] * inv);
  }
}

// ---------------------------------------------------------------------------
extern "C" void kernel_launch(void* const* d_in, const int* in_sizes, int n_in,
                              void* d_out, int out_size, void* d_ws, size_t ws_size,
                              hipStream_t stream) {
  (void)in_sizes; (void)n_in; (void)out_size; (void)ws_size;
  const float* x    = (const float*)d_in[0];
  const float* m    = (const float*)d_in[1];
  const float* Wqk  = (const float*)d_in[2];
  const float* mWqk = (const float*)d_in[3];
  const float* Wv   = (const float*)d_in[4];
  const float* mWv  = (const float*)d_in[5];
  const float* Wo   = (const float*)d_in[6];
  const float* bo   = (const float*)d_in[7];
  const float* mWo  = (const float*)d_in[8];
  const float* mbo  = (const float*)d_in[9];
  const float* Wf   = (const float*)d_in[10];
  const float* bf_  = (const float*)d_in[11];
  float* out = (float*)d_out;

  char* wp = (char*)d_ws;
  auto carve = [&](size_t elems) {
    bf16_t* r = (bf16_t*)wp;
    wp += ((elems * sizeof(bf16_t)) + 255) & ~(size_t)255;
    return r;
  };
  bf16_t* xb    = carve((size_t)MROWS * DM);
  bf16_t* mb    = carve((size_t)MROWS * DM);
  bf16_t* WqkT  = carve((size_t)INNER * DM);
  bf16_t* mWqkT = carve((size_t)INNER * DM);
  bf16_t* WvT   = carve((size_t)INNER * DM);
  bf16_t* mWvT  = carve((size_t)INNER * DM);
  bf16_t* WoT   = carve((size_t)DM * INNER);
  bf16_t* mWoT  = carve((size_t)DM * INNER);
  bf16_t* WfT   = carve((size_t)DM * DM);
  bf16_t* qk    = carve((size_t)MROWS * INNER);
  bf16_t* mqk   = carve((size_t)MROWS * INNER);
  bf16_t* vT    = carve((size_t)B_ * H * DH * SEQ);
  bf16_t* mvT   = carve((size_t)B_ * H * DH * SEQ);
  bf16_t* outH  = carve((size_t)MROWS * INNER);
  bf16_t* moutH = carve((size_t)MROWS * INNER);
  bf16_t* comb  = carve((size_t)CROWS * DM);

  // 1) fp32 -> bf16 activations; weight transposes
  {
    int n = MROWS * DM;
    f2bf_kernel<<<(n + 255) / 256, 256, 0, stream>>>(x, xb, n);
    f2bf_kernel<<<(n + 255) / 256, 256, 0, stream>>>(m, mb, n);
    int nw = DM * INNER;
    wtrans_kernel<<<(nw + 255) / 256, 256, 0, stream>>>(Wqk, WqkT, DM, INNER);
    wtrans_kernel<<<(nw + 255) / 256, 256, 0, stream>>>(mWqk, mWqkT, DM, INNER);
    wtrans_kernel<<<(nw + 255) / 256, 256, 0, stream>>>(Wv, WvT, DM, INNER);
    wtrans_kernel<<<(nw + 255) / 256, 256, 0, stream>>>(mWv, mWvT, DM, INNER);
    wtrans_kernel<<<(nw + 255) / 256, 256, 0, stream>>>(Wo, WoT, INNER, DM);
    wtrans_kernel<<<(nw + 255) / 256, 256, 0, stream>>>(mWo, mWoT, INNER, DM);
    int nf = DM * DM;
    wtrans_kernel<<<(nf + 255) / 256, 256, 0, stream>>>(Wf, WfT, DM, DM);
  }

  // 2) projections (M=4096, K=256): qk/mqk row-major, v/mv head-transposed
  {
    dim3 grid(INNER / 64, MROWS / 128), blk(256);
    gemm_bf16_kernel<0><<<grid, blk, 0, stream>>>(xb, WqkT, nullptr, qk, INNER, DM,
                                                  12, MROWS, 0);
    gemm_bf16_kernel<0><<<grid, blk, 0, stream>>>(mb, mWqkT, nullptr, mqk, INNER, DM,
                                                  12, MROWS, 0);
    gemm_bf16_kernel<1><<<grid, blk, 0, stream>>>(xb, WvT, nullptr, vT, INNER, DM,
                                                  0, 0, 0);
    gemm_bf16_kernel<1><<<grid, blk, 0, stream>>>(mb, mWvT, nullptr, mvT, INNER, DM,
                                                  0, 0, 0);
  }

  // 3) bidirectional attention = two flash passes (sim recomputed, never stored)
  {
    const float scale = 0.125f;  // DH^-0.5, DH = 64
    dim3 grid(SEQ / 128, B_ * H), blk(256);
    flash_attn_kernel<<<grid, blk, 0, stream>>>(qk, mqk, mvT, outH, scale);  // x -> m
    flash_attn_kernel<<<grid, blk, 0, stream>>>(mqk, qk, vT, moutH, scale);  // m -> x
  }

  // 4) output projections into combined buffer [b, 2*SEQ, DM] (bf16)
  {
    dim3 grid(DM / 64, MROWS / 128), blk(256);
    gemm_bf16_kernel<0><<<grid, blk, 0, stream>>>(outH, WoT, bo, comb, DM, INNER,
                                                  11, 2 * SEQ, 0);
    gemm_bf16_kernel<0><<<grid, blk, 0, stream>>>(moutH, mWoT, mbo, comb, DM, INNER,
                                                  11, 2 * SEQ, SEQ);
  }

  // 5) final projection -> fp32 output (M=8192, N=256, K=256)
  {
    dim3 grid(DM / 64, CROWS / 128), blk(256);
    gemm_bf16_kernel<2><<<grid, blk, 0, stream>>>(comb, WfT, bf_, out, DM, DM,
                                                  13, CROWS, 0);
  }
}
